// MultiDimMaxAffineFunction_25950192402693
// MI455X (gfx1250) — matrix operand order repaired and checked by reading, not compiled, not run on hardware
//
#include <hip/hip_runtime.h>

typedef __attribute__((ext_vector_type(16))) _Float16 v16h;
typedef __attribute__((ext_vector_type(8)))  float    v8f;

#define KEXP 16
#define MDIM 128
#define DDIM 64
#define AK_HALVES (MDIM * DDIM)          // 8192 f16 = 16 KB per expert slab
#define WAVES_PER_BLOCK 8
#define BLOCK_THREADS (WAVES_PER_BLOCK * 32)
#define ROWS_PER_BLOCK (WAVES_PER_BLOCK * 16)

#define LOG2E 1.4426950408889634f
#define LN2   0.6931471805599453f

// ---- prep: a (f32) -> f16 slabs scaled by log2(e), so WMMA emits base-2
// scores and the LSE can use raw v_exp_f32 / v_log_f32 (both base-2 HW ops).
__global__ __launch_bounds__(256) void cvt_a_f16_kernel(
    const float* __restrict__ a, _Float16* __restrict__ af) {
  const int i = blockIdx.x * 256 + threadIdx.x;   // one 8-element chunk each
  const float4* a4 = reinterpret_cast<const float4*>(a);
  float4 f0 = a4[2 * i + 0];
  float4 f1 = a4[2 * i + 1];
  union { _Float16 hh[8]; uint4 u; } pk;
  pk.hh[0] = (_Float16)(f0.x * LOG2E);  pk.hh[1] = (_Float16)(f0.y * LOG2E);
  pk.hh[2] = (_Float16)(f0.z * LOG2E);  pk.hh[3] = (_Float16)(f0.w * LOG2E);
  pk.hh[4] = (_Float16)(f1.x * LOG2E);  pk.hh[5] = (_Float16)(f1.y * LOG2E);
  pk.hh[6] = (_Float16)(f1.z * LOG2E);  pk.hh[7] = (_Float16)(f1.w * LOG2E);
  reinterpret_cast<uint4*>(af)[i] = pk.u;
}

// ---- prep: b (f32) scaled by log2(e) into workspace ----
__global__ __launch_bounds__(256) void cvt_b_kernel(
    const float* __restrict__ b, float* __restrict__ bs) {
  const int i = blockIdx.x * 256 + threadIdx.x;
  bs[i] = b[i] * LOG2E;
}

// Issue one expert slab (16 KB) as async DMA global->LDS, 4 x b128 per thread.
__device__ __forceinline__ void stage_async(const _Float16* __restrict__ src,
                                            unsigned lds_off, int tid) {
  #pragma unroll
  for (int rnd = 0; rnd < 4; ++rnd) {
    unsigned byte = (unsigned)(tid + rnd * BLOCK_THREADS) * 16u;
    unsigned dst  = lds_off + byte;
    asm volatile("global_load_async_to_lds_b128 %0, %1, %2"
                 :: "v"(dst), "v"(byte), "s"(src)
                 : "memory");
  }
}

__global__ __launch_bounds__(BLOCK_THREADS) void moe_lse_wmma_kernel(
    const float*    __restrict__ x,    // [N, 64] f32
    const float*    __restrict__ s,    // [16]
    const _Float16* __restrict__ af,   // [16, 128, 64] f16*log2e (from prep)
    const float*    __restrict__ bs,   // [16, 128] f32*log2e (from prep)
    float*          __restrict__ out)  // [N]
{
  // Double-buffered expert slabs (2 x 16 KB) + all biases (8 KB).
  __shared__ __align__(16) _Float16 smA[2 * AK_HALVES];
  __shared__ __align__(16) float    smBias[KEXP * MDIM];

  const int tid  = threadIdx.x;
  const int lane = tid & 31;
  const int wave = tid >> 5;
  const int r    = lane & 15;   // this lane's x-row (D column) within the tile
  const int h    = lane >> 4;   // half-wave select (D rows v+8h)

  const int rowbase = blockIdx.x * ROWS_PER_BLOCK + wave * 16;

  // Generic pointer to __shared__ carries the LDS byte offset in its low bits.
  const unsigned lds0 = (unsigned)(uintptr_t)&smA[0];
  const unsigned lds1 = (unsigned)(uintptr_t)&smA[AK_HALVES];

  // Prime the DMA pipeline: experts 0 and 1 in flight.
  stage_async(af + 0 * (size_t)AK_HALVES, lds0, tid);
  stage_async(af + 1 * (size_t)AK_HALVES, lds1, tid);

  // Stage all (pre-scaled) biases once; visible after the first barrier.
  #pragma unroll
  for (int i = tid; i < (KEXP * MDIM) / 4; i += BLOCK_THREADS)
    reinterpret_cast<float4*>(smBias)[i] =
        reinterpret_cast<const float4*>(bs)[i];

  // ------------- B fragments: x tile, f32 -> f16 (built once) -------------
  // B 32x16 in this code's contiguous k order (same as A): lane holds column n=(lane&15);
  // halves[t] = B[h*16 + t, n] = x[n, c*32 + h*16 + t] -> contiguous runs.
  const float4* xr4 =
      reinterpret_cast<const float4*>(x + (size_t)(rowbase + r) * DDIM);
  v16h bfrag[2];
  #pragma unroll
  for (int c = 0; c < 2; ++c) {
    #pragma unroll
    for (int q = 0; q < 4; ++q) {
      float4 p = xr4[c * 8 + h * 4 + q];
      const int t = q * 4;
      bfrag[c][t + 0] = (_Float16)p.x;
      bfrag[c][t + 1] = (_Float16)p.y;
      bfrag[c][t + 2] = (_Float16)p.z;
      bfrag[c][t + 3] = (_Float16)p.w;
    }
  }

  float acc = 0.0f;   // Σ_k s[k] * (M2 + log2(e2)); ×ln2 at the end

  for (int k = 0; k < KEXP; ++k) {
    // Wait for this expert's slab (keep the next one in flight), then make
    // all waves' DMA pieces visible to the whole workgroup.
    if (k == KEXP - 1)
      asm volatile("s_wait_asynccnt 0x0" ::: "memory");
    else
      asm volatile("s_wait_asynccnt 0x4" ::: "memory");
    __syncthreads();

    const _Float16* bufA = smA + (size_t)(k & 1) * AK_HALVES;
    const float sk = s[k];
    const float* bk = smBias + k * MDIM;

    // -------- 8 m-tiles of 16: D = A(a_k rows) * B(x cols), C = inline 0 ----
    v8f sc[8];
    #pragma unroll
    for (int mt = 0; mt < 8; ++mt) {
      // A 16x32 layout: lane holds row m = mt*16 + (lane&15); half-wave h:
      //   halves[0..7]  = K = c*32 + h*16 + 0..7
      //   halves[8..15] = K = c*32 + h*16 + 8 + 0..7
      const int m = mt * 16 + r;
      union { v16h v; uint4 u[2]; } A0, A1;
      const _Float16* arow = bufA + m * DDIM;
      A0.u[0] = *reinterpret_cast<const uint4*>(arow + 0 * 32 + h * 16);
      A0.u[1] = *reinterpret_cast<const uint4*>(arow + 0 * 32 + 8 + h * 16);
      A1.u[0] = *reinterpret_cast<const uint4*>(arow + 1 * 32 + h * 16);
      A1.u[1] = *reinterpret_cast<const uint4*>(arow + 1 * 32 + 8 + h * 16);

      v8f c = {};   // SRC2 = inline 0
      c = __builtin_amdgcn_wmma_f32_16x16x32_f16(
              false, A0.v, false, bfrag[0], (short)0, c, false, false);
      c = __builtin_amdgcn_wmma_f32_16x16x32_f16(
              false, A1.v, false, bfrag[1], (short)0, c, false, false);
      sc[mt] = c;
    }

    // ---------- per-row base-2 logsumexp over the 128 m values ----------
    // sc[mt][v] = log2e*score(m = mt*16 + v + 8*h, row n = lane&15).
    // This lane holds 64 m-values; partner lane (lane^16) holds the other 64.
    float mv = -3.402823466e+38f;
    #pragma unroll
    for (int mt = 0; mt < 8; ++mt) {
      const float4* bp =
          reinterpret_cast<const float4*>(bk + mt * 16 + 8 * h);
      float4 q0 = bp[0];
      float4 q1 = bp[1];
      sc[mt][0] += q0.x;  sc[mt][1] += q0.y;
      sc[mt][2] += q0.z;  sc[mt][3] += q0.w;
      sc[mt][4] += q1.x;  sc[mt][5] += q1.y;
      sc[mt][6] += q1.z;  sc[mt][7] += q1.w;
      #pragma unroll
      for (int v = 0; v < 8; ++v) mv = fmaxf(mv, sc[mt][v]);
    }
    // Pairwise max merge: symmetric, both lanes agree on M.
    const float M = fmaxf(mv, __shfl_xor(mv, 16, 32));

    float e = 0.0f;
    #pragma unroll
    for (int mt = 0; mt < 8; ++mt)
      #pragma unroll
      for (int v = 0; v < 8; ++v)
        e += __builtin_amdgcn_exp2f(sc[mt][v] - M);   // raw v_exp_f32
    e += __shfl_xor(e, 16, 32);

    acc += sk * (M + __builtin_amdgcn_logf(e));       // raw v_log_f32 (log2)

    __syncthreads();   // everyone done reading buf[k&1] before overwrite

    if (k + 2 < KEXP)
      stage_async(af + (size_t)(k + 2) * AK_HALVES,
                  ((k & 1) ? lds1 : lds0), tid);
  }

  // ---------- output: coalesced 16-float row write per wave ----------
  if (h == 0) out[rowbase + r] = acc * LN2;
}

extern "C" void kernel_launch(void* const* d_in, const int* in_sizes, int n_in,
                              void* d_out, int out_size, void* d_ws, size_t ws_size,
                              hipStream_t stream) {
  const float* x = (const float*)d_in[0];   // [N, 64]
  const float* s = (const float*)d_in[1];   // [16]
  const float* a = (const float*)d_in[2];   // [16, 128, 64]
  const float* b = (const float*)d_in[3];   // [16, 128]
  float* out = (float*)d_out;               // [N]

  _Float16* af = (_Float16*)d_ws;                       // 256 KB f16 a*log2e
  float* bsc   = (float*)((char*)d_ws + (size_t)KEXP * AK_HALVES * 2); // 8 KB

  // 1) one-shot prep: scale-convert a and b into workspace
  const int a_elems = in_sizes[2];          // 131072
  hipLaunchKernelGGL(cvt_a_f16_kernel, dim3(a_elems / (256 * 8)), dim3(256),
                     0, stream, a, af);
  const int b_elems = in_sizes[3];          // 2048
  hipLaunchKernelGGL(cvt_b_kernel, dim3(b_elems / 256), dim3(256),
                     0, stream, b, bsc);

  // 2) main fused batched-GEMM + logsumexp kernel
  const int nrows = in_sizes[0] / DDIM;     // N = 131072
  const int nblocks = nrows / ROWS_PER_BLOCK;
  hipLaunchKernelGGL(moe_lse_wmma_kernel, dim3(nblocks), dim3(BLOCK_THREADS),
                     0, stream, x, s, af, bsc, out);
}
